// FastDTW_40312563041029
// MI455X (gfx1250) — matrix operand order repaired and checked by reading, not compiled, not run on hardware
//
#include <hip/hip_runtime.h>
#include <hip/hip_bf16.h>
#include <math.h>

// DTW for B=8, N=M=2048, F=64 fp32 on MI455X (gfx1250, wave32).
//
// Stage 0 (prep):  split x,y into bf16 hi/lo pairs (bf16x3 scheme) and compute
//                  exact fp32 row norms (wave shuffle reduction). ~17 MB traffic.
// Stage 1 (dist):  dist[b,i,j] = sqrt(max(x2+y2-2*x.y,0)) via
//                  V_WMMA_F32_16X16X32_BF16; each wave holds one A fragment and
//                  streams 4 B tiles -> 24 WMMAs/wave, near-zero VALU overhead.
//                  134 MB dist stays resident in the 192 MB L2.
// Stage 2 (DTW):   anti-diagonal wavefront, 1 workgroup/batch, 2 rows/thread in
//                  registers, one LDS exchange + one barrier per step, with
//                  global_prefetch_b8 a cacheline ahead of the strided loads.

#define B_ 8
#define N_ 2048
#define M_ 2048
#define F_ 64

typedef __attribute__((ext_vector_type(16))) __bf16 v16bf;
typedef __attribute__((ext_vector_type(8)))  __bf16 v8bf;
typedef __attribute__((ext_vector_type(8)))  float  v8f;

union BfVec { v16bf v; __bf16 s[16]; };

// ---------------------------------------------------------------------------
// Stage 0: one wave per row (64 floats, 2/lane). Emits bf16 hi, bf16 residual,
// and the fp32 squared norm of the row.
__global__ void __launch_bounds__(256)
prep_kernel(const float* __restrict__ x, const float* __restrict__ y,
            __bf16* __restrict__ xhi, __bf16* __restrict__ xlo,
            __bf16* __restrict__ yhi, __bf16* __restrict__ ylo,
            float* __restrict__ x2g, float* __restrict__ y2g) {
  const int lane = threadIdx.x & 31;
  const int w    = threadIdx.x >> 5;
  const int row  = blockIdx.x * 8 + w;          // 0 .. B*N-1
  const bool isY = (blockIdx.y != 0);

  const float* src = (isY ? y : x) + (size_t)row * F_;
  __bf16* hi = (isY ? yhi : xhi) + (size_t)row * F_;
  __bf16* lo = (isY ? ylo : xlo) + (size_t)row * F_;

  float2 f = *reinterpret_cast<const float2*>(src + 2 * lane);
  __bf16 h0 = (__bf16)f.x, h1 = (__bf16)f.y;
  __bf16 l0 = (__bf16)(f.x - (float)h0);
  __bf16 l1 = (__bf16)(f.y - (float)h1);
  union { __bf16 b[2]; unsigned u; } ph, pl;
  ph.b[0] = h0; ph.b[1] = h1;
  pl.b[0] = l0; pl.b[1] = l1;
  *reinterpret_cast<unsigned*>(hi + 2 * lane) = ph.u;
  *reinterpret_cast<unsigned*>(lo + 2 * lane) = pl.u;

  float s = f.x * f.x + f.y * f.y;
#pragma unroll
  for (int m = 16; m >= 1; m >>= 1) s += __shfl_xor(s, m, 32);
  if (lane == 0) (isY ? y2g : x2g)[row] = s;
}

// ---------------------------------------------------------------------------
// Stage 1: one wave = 16(i) x 64(j) of dist; block (8 waves) = 16 x 512.
// grid = (M/512, N/16, B), block = 256.
//
// A fragment (16x32 bf16, ISA 7.12.2): lane = row (lane&15), half = lane>>4,
//   elem e<8 -> K = half*8+e ; e>=8 -> K = 16+half*8+(e-8)   (two B128 loads)
// B fragment (32x16 bf16): lane = col (lane&15), same K mapping as A
//   (two B128 loads)
__global__ void __launch_bounds__(256)
dist_wmma_kernel(const __bf16* __restrict__ xhi, const __bf16* __restrict__ xlo,
                 const __bf16* __restrict__ yhi, const __bf16* __restrict__ ylo,
                 const float* __restrict__ x2g, const float* __restrict__ y2g,
                 float* __restrict__ dist) {
  __shared__ float xs[16];

  const int lane = threadIdx.x & 31;
  const int w    = threadIdx.x >> 5;
  const int b    = blockIdx.z;
  const int i0   = blockIdx.y * 16;
  const int j0   = blockIdx.x * 512 + w * 64;

  if (threadIdx.x < 16) xs[threadIdx.x] = x2g[b * N_ + i0 + threadIdx.x];
  __syncthreads();

  const int n    = lane & 15;
  const int half = lane >> 4;

  const __bf16* xh = xhi + ((size_t)b * N_ + i0 + n) * F_;
  const __bf16* xl = xlo + ((size_t)b * N_ + i0 + n) * F_;

  v8f acc0 = {}, acc1 = {}, acc2 = {}, acc3 = {};
#pragma unroll
  for (int c = 0; c < 2; ++c) {                 // two K=32 chunks
    const int kb1 = c * 32 + half * 8;          // A pattern
    const int kb2 = kb1 + 16;
    const int kbB = c * 32 + half * 8;          // B pattern

    BfVec ah, al;                                // A fragment: loaded once,
    *(v8bf*)&ah.s[0] = *(const v8bf*)(xh + kb1); // reused for 4 B tiles
    *(v8bf*)&ah.s[8] = *(const v8bf*)(xh + kb2);
    *(v8bf*)&al.s[0] = *(const v8bf*)(xl + kb1);
    *(v8bf*)&al.s[8] = *(const v8bf*)(xl + kb2);

#pragma unroll
    for (int tt = 0; tt < 4; ++tt) {
      const size_t yrow = ((size_t)b * M_ + j0 + tt * 16 + n) * F_;
      BfVec bh, bl;
      *(v8bf*)&bh.s[0] = *(const v8bf*)(yhi + yrow + kbB);
      *(v8bf*)&bh.s[8] = *(const v8bf*)(yhi + yrow + kbB + 16);
      *(v8bf*)&bl.s[0] = *(const v8bf*)(ylo + yrow + kbB);
      *(v8bf*)&bl.s[8] = *(const v8bf*)(ylo + yrow + kbB + 16);

      v8f& a = (tt == 0) ? acc0 : (tt == 1) ? acc1 : (tt == 2) ? acc2 : acc3;
      // bf16x3: hi*hi + lo*hi + hi*lo (lo*lo dropped), fp32 accumulate
      a = __builtin_amdgcn_wmma_f32_16x16x32_bf16(false, ah.v, false, bh.v,
                                                  (short)0, a, false, false);
      a = __builtin_amdgcn_wmma_f32_16x16x32_bf16(false, al.v, false, bh.v,
                                                  (short)0, a, false, false);
      a = __builtin_amdgcn_wmma_f32_16x16x32_bf16(false, ah.v, false, bl.v,
                                                  (short)0, a, false, false);
    }
  }

  // C/D layout: lane = column n; VGPR r holds row m = r + 8*half.
#pragma unroll
  for (int tt = 0; tt < 4; ++tt) {
    const v8f& a = (tt == 0) ? acc0 : (tt == 1) ? acc1 : (tt == 2) ? acc2 : acc3;
    const int j = j0 + tt * 16 + n;
    const float y2 = y2g[b * M_ + j];
    float* dout = dist + ((size_t)b * N_ + i0) * M_ + j;
#pragma unroll
    for (int r = 0; r < 8; ++r) {
      const int m = r + 8 * half;
      float d2 = xs[m] + y2 - 2.0f * a[r];
      dout[(size_t)m * M_] = sqrtf(fmaxf(d2, 0.0f));
    }
  }
}

// ---------------------------------------------------------------------------
// Stage 2: wavefront DP. grid = B, block = 1024; thread t owns rows 2t, 2t+1.
__global__ void __launch_bounds__(1024)
dtw_wavefront_kernel(const float* __restrict__ dist, float* __restrict__ out) {
  __shared__ float ex1[2][1024];
  __shared__ float ex2[2][1024];

  const int t = threadIdx.x;
  const int b = blockIdx.x;
  const float INF = __builtin_inff();
  const size_t base = (size_t)b * N_ * M_;
  const int i0 = 2 * t, i1 = 2 * t + 1;

  float p1a = INF, p1b = INF;   // diagonal d-1, rows i0/i1
  float p2a = INF, p2b = INF;   // diagonal d-2
  int buf = 0;

  for (int d = 0; d < N_ + M_ - 1; ++d) {
    const int j0 = d - i0, j1 = d - i1;
    const bool v0 = (j0 >= 0) & (j0 < M_);
    const bool v1 = (j1 >= 0) & (j1 < M_);
    const int jc0 = min(max(j0, 0), M_ - 1);
    const int jc1 = min(max(j1, 0), M_ - 1);
    const float* a0 = dist + base + (size_t)i0 * M_ + jc0;
    const float* a1 = dist + base + (size_t)i1 * M_ + jc1;
    const float d0 = *a0;
    const float d1 = *a1;
    // Pull the next cacheline of each row toward the WGP while we sync.
    __builtin_prefetch(a0 + 32, 0, 1);
    __builtin_prefetch(a1 + 32, 0, 1);

    ex1[buf][t] = p1b;          // row 2t+1 on diagonal d-1
    ex2[buf][t] = p2b;          // row 2t+1 on diagonal d-2
    __syncthreads();
    const float n1 = (t > 0) ? ex1[buf][t - 1] : INF;  // prev1[i0-1]
    const float n2 = (t > 0) ? ex2[buf][t - 1] : INF;  // prev2[i0-1]

    float cand0 = fminf(p1a, fminf(n1, n2));
    if (d == 0 && t == 0) cand0 = 0.0f;                // D[0,0] boundary
    const float cand1 = fminf(p1b, fminf(p1a, p2a));   // thread-local

    const float cur0 = v0 ? (d0 + cand0) : INF;
    const float cur1 = v1 ? (d1 + cand1) : INF;

    p2a = p1a; p2b = p1b;
    p1a = cur0; p1b = cur1;
    buf ^= 1;
  }

  if (t == 1023) out[b] = p1b;  // D[N-1, M-1]
}

// ---------------------------------------------------------------------------
extern "C" void kernel_launch(void* const* d_in, const int* in_sizes, int n_in,
                              void* d_out, int out_size, void* d_ws, size_t ws_size,
                              hipStream_t stream) {
  const float* x = (const float*)d_in[0];  // [B, N, F] fp32
  const float* y = (const float*)d_in[1];  // [B, M, F] fp32
  float* out = (float*)d_out;              // [B] fp32

  // Workspace layout (needs ~136.2 MiB):
  char* p = (char*)d_ws;
  float*  dist = (float*)p;  p += (size_t)B_ * N_ * M_ * sizeof(float);  // 134 MB
  __bf16* xhi  = (__bf16*)p; p += (size_t)B_ * N_ * F_ * sizeof(__bf16); // 2 MB
  __bf16* xlo  = (__bf16*)p; p += (size_t)B_ * N_ * F_ * sizeof(__bf16);
  __bf16* yhi  = (__bf16*)p; p += (size_t)B_ * M_ * F_ * sizeof(__bf16);
  __bf16* ylo  = (__bf16*)p; p += (size_t)B_ * M_ * F_ * sizeof(__bf16);
  float*  x2g  = (float*)p;  p += (size_t)B_ * N_ * sizeof(float);       // 64 KB
  float*  y2g  = (float*)p;

  prep_kernel<<<dim3((B_ * N_) / 8, 2), dim3(256), 0, stream>>>(
      x, y, xhi, xlo, yhi, ylo, x2g, y2g);

  dist_wmma_kernel<<<dim3(M_ / 512, N_ / 16, B_), dim3(256), 0, stream>>>(
      xhi, xlo, yhi, ylo, x2g, y2g, dist);

  dtw_wavefront_kernel<<<dim3(B_), dim3(1024), 0, stream>>>(dist, out);
}
